// ScaledTransformerBlock_30313879175871
// MI455X (gfx1250) — compile-verified, matrix operand order checked by reading, not run on hardware
//
#include <hip/hip_runtime.h>
#include <hip/hip_bf16.h>

typedef __attribute__((ext_vector_type(16))) __bf16        v16bf;
typedef __attribute__((ext_vector_type(8)))  float         v8f;
typedef __attribute__((ext_vector_type(4)))  unsigned int  u32x4;
typedef __attribute__((ext_vector_type(8)))  int           i32x8;
typedef __attribute__((ext_vector_type(4)))  int           i32x4;

// ---------------------------------------------------------------------------
// Tensor Data Mover: DMA a 2D bf16 tile [tile_rows x tile_k] (row stride
// row_stride elems) from global memory into LDS at lds_byte_off, padding
// 8 DWORDs after every 16 DWORDs (one 32-elem bf16 row) -> LDS pitch 48 elems.
// D# layout per CDNA5 ISA ch.8 (group0 128b, group1 256b; groups 2/3 zero).
// ---------------------------------------------------------------------------
__device__ __forceinline__ void tdm_load_tile_bf16(const __bf16* gptr,
                                                   unsigned lds_byte_off,
                                                   int tile_k, int tile_rows,
                                                   long row_stride) {
  const unsigned long long ga = (unsigned long long)(size_t)(const void*)gptr;
  u32x4 g0;
  g0[0] = 1u;                                        // count=1, user descriptor
  g0[1] = lds_byte_off;                              // lds_addr
  g0[2] = (unsigned)(ga & 0xffffffffu);              // global_addr[31:0]
  g0[3] = (unsigned)((ga >> 32) & 0x1ffffffu) | (2u << 30);  // ga[56:32], type=2
  i32x8 g1;
  g1[0] = (1 << 16)        // data_size = 2 bytes
        | (1 << 20)        // pad_enable
        | (3 << 22)        // pad_interval: 16 DWORDs (64B row)
        | (7 << 25);       // pad_amount:   8 DWORDs (32B) -> pitch 96B
  g1[1] = (tile_k & 0xffff) << 16;                   // tensor_dim0[15:0]
  g1[2] = ((tile_k >> 16) & 0xffff)                  // tensor_dim0[31:16]
        | ((tile_rows & 0xffff) << 16);              // tensor_dim1[15:0]
  g1[3] = ((tile_rows >> 16) & 0xffff)               // tensor_dim1[31:16]
        | ((tile_k & 0xffff) << 16);                 // tile_dim0
  g1[4] = (tile_rows & 0xffff);                      // tile_dim1 (tile_dim2=0)
  g1[5] = (int)(row_stride & 0xffffffffL);           // tensor_dim0_stride[31:0]
  g1[6] = (int)((row_stride >> 32) & 0xffffL);       // stride[47:32]; dim1_stride=0
  g1[7] = 0;
  const i32x4 gz = {0, 0, 0, 0};
#if defined(__clang_major__) && (__clang_major__ >= 23)
  const i32x8 gz8 = {0, 0, 0, 0, 0, 0, 0, 0};
  __builtin_amdgcn_tensor_load_to_lds(g0, g1, gz, gz, gz8, 0);
#else
  __builtin_amdgcn_tensor_load_to_lds(g0, g1, gz, gz, 0);
#endif
}

// ---------------------------------------------------------------------------
// Elementwise fp32 -> bf16 cast (weights)
// ---------------------------------------------------------------------------
__global__ __launch_bounds__(256) void cast_f32_to_bf16(const float* __restrict__ in,
                                                        __bf16* __restrict__ out, long n) {
  long i = (long)blockIdx.x * blockDim.x + threadIdx.x;
  long stride = (long)gridDim.x * blockDim.x;
  for (; i < n; i += stride) out[i] = (__bf16)in[i];
}

// ---------------------------------------------------------------------------
// LayerNorm over D=1024, one row per block (256 threads x 4 elems), bf16 out
// ---------------------------------------------------------------------------
__global__ __launch_bounds__(256) void layernorm_to_bf16(const float* __restrict__ x,
                                                         const float* __restrict__ g,
                                                         const float* __restrict__ b,
                                                         __bf16* __restrict__ out) {
  __shared__ float s1[256];
  __shared__ float s2[256];
  const int row = blockIdx.x;
  const int t = threadIdx.x;
  const float* xr = x + (long)row * 1024;
  float4 v = *(const float4*)(xr + t * 4);
  s1[t] = v.x + v.y + v.z + v.w;
  s2[t] = v.x * v.x + v.y * v.y + v.z * v.z + v.w * v.w;
  __syncthreads();
  for (int o = 128; o > 0; o >>= 1) {
    if (t < o) { s1[t] += s1[t + o]; s2[t] += s2[t + o]; }
    __syncthreads();
  }
  const float mu = s1[0] * (1.0f / 1024.0f);
  const float var = s2[0] * (1.0f / 1024.0f) - mu * mu;
  const float rs = rsqrtf(var + 1e-5f);
  const float4 gg = *(const float4*)(g + t * 4);
  const float4 bb = *(const float4*)(b + t * 4);
  __bf16* orow = out + (long)row * 1024;
  orow[t * 4 + 0] = (__bf16)((v.x - mu) * rs * gg.x + bb.x);
  orow[t * 4 + 1] = (__bf16)((v.y - mu) * rs * gg.y + bb.y);
  orow[t * 4 + 2] = (__bf16)((v.z - mu) * rs * gg.z + bb.z);
  orow[t * 4 + 3] = (__bf16)((v.w - mu) * rs * gg.w + bb.w);
}

// ---------------------------------------------------------------------------
// Masked softmax over rows of 2048, in place on the attention-weights output.
// ---------------------------------------------------------------------------
__global__ __launch_bounds__(256) void softmax_mask_inplace(float* __restrict__ attn,
                                                            const int* __restrict__ mask) {
  __shared__ float red[256];
  const long row = blockIdx.x;
  float* p = attn + row * 2048L;
  const int* m = mask + (long)(row & 2047) * 2048;
  const int t = threadIdx.x;
  float v[8];
#pragma unroll
  for (int j = 0; j < 8; ++j) {
    const int idx = t + j * 256;
    const float s = p[idx];
    v[j] = (m[idx] == 0) ? -__builtin_inff() : s;
  }
  float mx = v[0];
#pragma unroll
  for (int j = 1; j < 8; ++j) mx = fmaxf(mx, v[j]);
  red[t] = mx;
  __syncthreads();
  for (int o = 128; o > 0; o >>= 1) {
    if (t < o) red[t] = fmaxf(red[t], red[t + o]);
    __syncthreads();
  }
  mx = red[0];
  __syncthreads();
  float sum = 0.0f;
#pragma unroll
  for (int j = 0; j < 8; ++j) { v[j] = __expf(v[j] - mx); sum += v[j]; }
  red[t] = sum;
  __syncthreads();
  for (int o = 128; o > 0; o >>= 1) {
    if (t < o) red[t] += red[t + o];
    __syncthreads();
  }
  const float inv = 1.0f / red[0];
#pragma unroll
  for (int j = 0; j < 8; ++j) p[t + j * 256] = v[j] * inv;
}

// ---------------------------------------------------------------------------
// Generic batched WMMA GEMM:  C = epilogue(alpha * A @ B + bias [+resid])
//   A: bf16 row-major (TDM-staged), or fp32 converted while staging (AF32)
//   B: bf16 row-major KxN (manual transpose into LDS), or BTRANS: element
//      (k,n) at offB + n*ldb + k (TDM-staged)
// Block tile 128 x BN, BK=32, 256 threads = 8 wave32.
// Double-buffered LDS pipeline: tile t+1 is DMA'd (TDM) / reg-staged while
// tile t feeds the WMMA chain; s_wait_tensorcnt(NTDM) drains only the older
// DMAs, leaving the new ones in flight.
// ---------------------------------------------------------------------------
template <int BN, bool AF32, bool BTRANS, bool OUTBF, bool HASBIAS, bool DOGELU, bool DORESID>
__global__ __launch_bounds__(256) void wmma_gemm_bf16(
    const void* __restrict__ A, const __bf16* __restrict__ B, void* __restrict__ C,
    const float* __restrict__ bias, const float* __restrict__ resid,
    int M, int N, int K, long lda, long ldb, long ldc, float alpha,
    int bdiv, long a_so, long a_si, long b_so, long b_si, long c_so, long c_si) {
  constexpr int BM = 128, BK = 32, LP = 48;  // LDS pitch 48 elems = 96 B
  constexpr int TILES = BN / 16;
  constexpr int NTDM = (!AF32 ? 1 : 0) + (BTRANS ? 1 : 0);
  constexpr int AIT = (BM * BK) / (256 * 4);  // fp32 A staging chunks (4)
  constexpr int BIT = (BN * BK) / (256 * 8);  // manual B staging chunks
  __shared__ __bf16 As[2][BM * LP];
  __shared__ __bf16 Bs[2][BN * LP];  // stored transposed: Bs[n][kk]

  const int z = blockIdx.z;
  const long offA = (long)(z / bdiv) * a_so + (long)(z % bdiv) * a_si;
  const long offB = (long)(z / bdiv) * b_so + (long)(z % bdiv) * b_si;
  const long offC = (long)(z / bdiv) * c_so + (long)(z % bdiv) * c_si;

  const int rowBase = blockIdx.y * BM;
  const int colBase = blockIdx.x * BN;
  const int tid = threadIdx.x;
  const int wave = tid >> 5;
  const int lane = tid & 31;
  const int lhi = lane >> 4;
  const int llo = lane & 15;

  const unsigned asLds = (unsigned)(size_t)(const void*)&As[0][0];
  const unsigned bsLds = (unsigned)(size_t)(const void*)&Bs[0][0];

  float4 aReg[AIT];  // next-tile staging registers (AF32 path)
  float4 bReg[BIT];  // next-tile staging registers (manual-B path)

  auto issueTDM = [&](int t, int buf) {
    if constexpr (!AF32)
      tdm_load_tile_bf16((const __bf16*)A + offA + (long)rowBase * lda + (long)t * BK,
                         asLds + (unsigned)buf * (BM * LP * 2), BK, BM, lda);
    if constexpr (BTRANS)
      tdm_load_tile_bf16(B + offB + (long)colBase * ldb + (long)t * BK,
                         bsLds + (unsigned)buf * (BN * LP * 2), BK, BN, ldb);
  };
  auto loadA = [&](int t) {  // AF32: global fp32 -> registers
    const float* Af = (const float*)A + offA;
#pragma unroll
    for (int it = 0; it < AIT; ++it) {
      const int i = it * 1024 + tid * 4;
      const int r = i / BK, c = i % BK;
      aReg[it] = *(const float4*)(Af + (long)(rowBase + r) * lda + ((long)t * BK + c));
    }
  };
  auto storeA = [&](int buf) {  // registers -> LDS (bf16 convert)
#pragma unroll
    for (int it = 0; it < AIT; ++it) {
      const int i = it * 1024 + tid * 4;
      const int r = i / BK, c = i % BK;
      __bf16* d = &As[buf][r * LP + c];
      d[0] = (__bf16)aReg[it].x; d[1] = (__bf16)aReg[it].y;
      d[2] = (__bf16)aReg[it].z; d[3] = (__bf16)aReg[it].w;
    }
  };
  auto loadB = [&](int t) {  // manual B: global bf16 -> registers
    const __bf16* Bb = B + offB;
#pragma unroll
    for (int it = 0; it < BIT; ++it) {
      const int i = it * 2048 + tid * 8;
      const int kk = i / BN, n0 = i % BN;
      bReg[it] = *(const float4*)(Bb + (long)((long)t * BK + kk) * ldb + (colBase + n0));
    }
  };
  auto storeB = [&](int buf) {  // registers -> LDS (transposed scatter)
#pragma unroll
    for (int it = 0; it < BIT; ++it) {
      const int i = it * 2048 + tid * 8;
      const int kk = i / BN, n0 = i % BN;
      const __bf16* pv = (const __bf16*)&bReg[it];
#pragma unroll
      for (int j = 0; j < 8; ++j) Bs[buf][(n0 + j) * LP + kk] = pv[j];
    }
  };

  v8f acc[TILES] = {};
  const int T = K / BK;

  // ---- prologue: stage tile 0 into buffer 0 ----
  if (wave == 0) issueTDM(0, 0);
  if constexpr (AF32) { loadA(0); storeA(0); }
  if constexpr (!BTRANS) { loadB(0); storeB(0); }

  for (int t = 0; t < T; ++t) {
    const int buf = t & 1, nbuf = buf ^ 1;
    const bool more = (t + 1) < T;
    // ---- kick off tile t+1 into the other buffer (last read at t-1) ----
    if (more) {
      if (wave == 0) issueTDM(t + 1, nbuf);
      if constexpr (AF32) loadA(t + 1);
      if constexpr (!BTRANS) loadB(t + 1);
    }
    // ---- drain only the OLDER (tile t) DMAs; tile t+1's stay in flight ----
    if constexpr (NTDM > 0) {
      if (wave == 0) {
        if (more) __builtin_amdgcn_s_wait_tensorcnt(NTDM);
        else      __builtin_amdgcn_s_wait_tensorcnt(0);
      }
    }
    if (more) {
      if constexpr (AF32) storeA(nbuf);
      if constexpr (!BTRANS) storeB(nbuf);
    }
    __syncthreads();  // tile t (TDM + manual stores from iter t-1) visible

    // ---- compute tile t ----
    const v16bf afrag = *(const v16bf*)&As[buf][(wave * 16 + llo) * LP + lhi * 16];
    v16bf bfr[TILES];
#pragma unroll
    for (int u = 0; u < TILES; ++u)
      bfr[u] = *(const v16bf*)&Bs[buf][(u * 16 + llo) * LP + lhi * 16];
#pragma unroll
    for (int u = 0; u < TILES; ++u)
      acc[u] = __builtin_amdgcn_wmma_f32_16x16x32_bf16(false, afrag, false, bfr[u],
                                                       (short)0, acc[u], false, false);
    __syncthreads();  // licenses next iteration's writes into 'buf'
  }

  // ---- epilogue: C/D layout = VGPR r -> (M = r + 8*lhi, N = llo) ----
  const int rowW = rowBase + wave * 16 + lhi * 8;
#pragma unroll
  for (int t = 0; t < TILES; ++t) {
    const int col = colBase + t * 16 + llo;
#pragma unroll
    for (int r = 0; r < 8; ++r) {
      float v = acc[t][r] * alpha;
      if constexpr (HASBIAS) v += bias[col];
      if constexpr (DOGELU) v = 0.5f * v * (1.0f + erff(v * 0.70710678118654752f));
      const long ci = offC + (long)(rowW + r) * ldc + col;
      if constexpr (DORESID) v += resid[ci];
      if constexpr (OUTBF) ((__bf16*)C)[ci] = (__bf16)v;
      else                 ((float*)C)[ci] = v;
    }
  }
}

// ---------------------------------------------------------------------------
// Launcher
// ---------------------------------------------------------------------------
extern "C" void kernel_launch(void* const* d_in, const int* in_sizes, int n_in,
                              void* d_out, int out_size, void* d_ws, size_t ws_size,
                              hipStream_t stream) {
  (void)in_sizes; (void)n_in; (void)out_size; (void)ws_size;
  const float* x     = (const float*)d_in[0];
  const int*   mask  = (const int*)d_in[1];
  const float* Wqkv  = (const float*)d_in[2];
  const float* bqkv  = (const float*)d_in[3];
  const float* Wproj = (const float*)d_in[4];
  const float* bproj = (const float*)d_in[5];
  const float* ln1g  = (const float*)d_in[6];
  const float* ln1b  = (const float*)d_in[7];
  const float* ln2g  = (const float*)d_in[8];
  const float* ln2b  = (const float*)d_in[9];
  const float* W1    = (const float*)d_in[10];
  const float* b1    = (const float*)d_in[11];
  const float* W2    = (const float*)d_in[12];
  const float* b2    = (const float*)d_in[13];

  const long BS = 2L * 2048;  // B*S = 4096 rows
  float* out_x    = (float*)d_out;              // [4096, 1024]
  float* out_attn = (float*)d_out + BS * 1024;  // [32, 2048, 2048]

  char* ws = (char*)d_ws;
  size_t off = 0;
  auto carve = [&](size_t bytes) -> void* {
    void* p = ws + off;
    off += (bytes + 255) & ~(size_t)255;
    return p;
  };
  __bf16* hbf    = (__bf16*)carve((size_t)BS * 1024 * 2);
  __bf16* qkvbf  = (__bf16*)carve((size_t)BS * 3072 * 2);
  __bf16* mrgbf  = (__bf16*)carve((size_t)BS * 1024 * 2);
  float*  x1     = (float*) carve((size_t)BS * 1024 * 4);
  __bf16* h2bf   = (__bf16*)carve((size_t)BS * 1024 * 2);
  __bf16* ffbf   = (__bf16*)carve((size_t)BS * 4096 * 2);
  __bf16* Wqkvb  = (__bf16*)carve((size_t)1024 * 3072 * 2);
  __bf16* Wprojb = (__bf16*)carve((size_t)1024 * 1024 * 2);
  __bf16* W1b    = (__bf16*)carve((size_t)1024 * 4096 * 2);
  __bf16* W2b    = (__bf16*)carve((size_t)4096 * 1024 * 2);

  cast_f32_to_bf16<<<512, 256, 0, stream>>>(Wqkv, Wqkvb, 1024L * 3072);
  cast_f32_to_bf16<<<512, 256, 0, stream>>>(Wproj, Wprojb, 1024L * 1024);
  cast_f32_to_bf16<<<512, 256, 0, stream>>>(W1, W1b, 1024L * 4096);
  cast_f32_to_bf16<<<512, 256, 0, stream>>>(W2, W2b, 4096L * 1024);

  layernorm_to_bf16<<<BS, 256, 0, stream>>>(x, ln1g, ln1b, hbf);

  // qkv = h @ Wqkv + bqkv   (bf16 out, [4096,3072]); A via TDM
  wmma_gemm_bf16<128, false, false, true, true, false, false>
      <<<dim3(3072 / 128, 4096 / 128, 1), 256, 0, stream>>>(
          hbf, Wqkvb, qkvbf, bqkv, nullptr, 4096, 3072, 1024, 1024, 3072, 3072,
          1.0f, 1, 0, 0, 0, 0, 0, 0);

  // scores[z] = (Q_z @ K_z^T)/8, z = b*16+h; A and B both via TDM
  wmma_gemm_bf16<128, false, true, false, false, false, false>
      <<<dim3(2048 / 128, 2048 / 128, 32), 256, 0, stream>>>(
          qkvbf, qkvbf + 1024, out_attn, nullptr, nullptr, 2048, 2048, 64,
          3072, 3072, 2048, 0.125f,
          16, 2048L * 3072, 64, 2048L * 3072, 64, 16L * 2048 * 2048, 2048L * 2048);

  softmax_mask_inplace<<<32 * 2048, 256, 0, stream>>>(out_attn, mask);

  // attn_out[z] = P_z @ V_z ; P fp32 converted at LDS stage; heads merged
  wmma_gemm_bf16<64, true, false, true, false, false, false>
      <<<dim3(1, 2048 / 128, 32), 256, 0, stream>>>(
          out_attn, qkvbf + 2048, mrgbf, nullptr, nullptr, 2048, 64, 2048,
          2048, 3072, 1024, 1.0f,
          16, 16L * 2048 * 2048, 2048L * 2048, 2048L * 3072, 64, 2048L * 1024, 64);

  // x1 = x + (attn_merged @ Wproj + bproj)
  wmma_gemm_bf16<128, false, false, false, true, false, true>
      <<<dim3(1024 / 128, 4096 / 128, 1), 256, 0, stream>>>(
          mrgbf, Wprojb, x1, bproj, x, 4096, 1024, 1024, 1024, 1024, 1024,
          1.0f, 1, 0, 0, 0, 0, 0, 0);

  layernorm_to_bf16<<<BS, 256, 0, stream>>>(x1, ln2g, ln2b, h2bf);

  // ff = gelu(h2 @ W1 + b1)
  wmma_gemm_bf16<128, false, false, true, true, true, false>
      <<<dim3(4096 / 128, 4096 / 128, 1), 256, 0, stream>>>(
          h2bf, W1b, ffbf, b1, nullptr, 4096, 4096, 1024, 1024, 4096, 4096,
          1.0f, 1, 0, 0, 0, 0, 0, 0);

  // out_x = x1 + (ff @ W2 + b2)
  wmma_gemm_bf16<128, false, false, false, true, false, true>
      <<<dim3(1024 / 128, 4096 / 128, 1), 256, 0, stream>>>(
          ffbf, W2b, out_x, b2, x1, 4096, 1024, 4096, 4096, 1024, 1024,
          1.0f, 1, 0, 0, 0, 0, 0, 0);
}
